// UNet1d_64_3layer_26474178412752
// MI455X (gfx1250) — matrix operand order repaired and checked by reading, not compiled, not run on hardware
//
#include <hip/hip_runtime.h>
#include <cstddef>
#include <cstdint>

// ---------------------------------------------------------------------------
// Types for CDNA5 WMMA (wave32, 16x16x32 bf16 -> f32)
// ---------------------------------------------------------------------------
typedef __bf16 v16bf __attribute__((ext_vector_type(16)));
typedef float  v8f   __attribute__((ext_vector_type(8)));

#define ACT_NONE     0
#define ACT_RELU     1
#define ACT_LRELU    2
#define ACT_SILU     3
#define ACT_SOFTPLUS 4

__device__ __forceinline__ float apply_act(float v, int act) {
  switch (act) {
    case ACT_RELU:     return v > 0.f ? v : 0.f;
    case ACT_LRELU:    return v > 0.f ? v : 0.2f * v;
    case ACT_SILU:     return v / (1.f + __expf(-v));
    case ACT_SOFTPLUS: return v > 20.f ? v : log1pf(__expf(v));
    default:           return v;
  }
}

// pack 4 f32 -> 4 bf16 and store 8B to LDS (v_cvt_pk_bf16_f32 + ds_store_b64)
__device__ __forceinline__ void store_pk4(__bf16* p, float4 v) {
  union { __bf16 h[4]; uint2 u; } t;
  t.h[0] = (__bf16)v.x; t.h[1] = (__bf16)v.y;
  t.h[2] = (__bf16)v.z; t.h[3] = (__bf16)v.w;
  *(uint2*)p = t.u;
}

// ---------------------------------------------------------------------------
// bf16 WMMA GEMM:  C[M,N] = act(A[M,K] @ W[N,K]^T + bias) (+ residual)
// A row-major, row stride lda (lda%4==0, 16B-aligned base); W row-major (N,K).
// Block: 256 threads = 8 waves; block tile 128x64, BK=64.
// Wave w: mt = w>>2 (64-row strip: 4 M-frags), nt = w&3 (16-col strip).
// 8 WMMAs per wave per staged tile (4 M-frags x 2 K-substeps, B frag reused).
// Interior tiles take a uniform-branch fast path (no per-element guards).
// ---------------------------------------------------------------------------
#define BM 128
#define BN 64
#define BK 64

__global__ __launch_bounds__(256)
void gemm_bf16_wmma(const float* __restrict__ A, int lda,
                    const float* __restrict__ W,
                    const float* __restrict__ bias,
                    float* __restrict__ C, int ldc,
                    const float* __restrict__ res,
                    int M, int N, int K, int act)
{
  __shared__ __align__(16) __bf16 As[BM][BK + 8];   // rows of 144B (16B multiple)
  __shared__ __align__(16) __bf16 Bs[BN][BK + 8];

  const int tid  = threadIdx.x;
  const int lane = tid & 31;
  const int wave = tid >> 5;
  const int mt   = wave >> 2;       // 0..1 : 64-row strip
  const int nt   = wave & 3;        // 0..3 : 16-col strip
  const int blockM = blockIdx.y * BM;
  const int blockN = blockIdx.x * BN;

  v8f acc[4];
  #pragma unroll
  for (int f = 0; f < 4; ++f) acc[f] = (v8f){0.f,0.f,0.f,0.f,0.f,0.f,0.f,0.f};

  const int srow = tid >> 4;        // staging row base 0..15
  const int skq  = tid & 15;        // float4 index within row

  const bool mnfull = (blockM + BM <= M) && (blockN + BN <= N);

  for (int kk = 0; kk < K; kk += BK) {
    if (kk + BK < K) {
      int pr = blockM + (tid & 127);
      if (pr < M) __builtin_prefetch(&A[(size_t)pr * lda + kk + BK], 0, 1);
    }

    if (mnfull && (kk + BK <= K)) {
      // ---------------- fast path: uniform, fully vectorized ----------------
      #pragma unroll
      for (int i = 0; i < 8; ++i) {          // A: 128 rows x 16 float4
        int row = srow + i * 16;
        float4 v = *(const float4*)(A + (size_t)(blockM + row) * lda + kk + skq * 4);
        store_pk4(&As[row][skq * 4], v);
      }
      #pragma unroll
      for (int i = 0; i < 4; ++i) {          // B: 64 rows x 16 float4
        int n = srow + i * 16;
        float4 v = *(const float4*)(W + (size_t)(blockN + n) * K + kk + skq * 4);
        store_pk4(&Bs[n][skq * 4], v);
      }
    } else {
      // ---------------- edge path: element guards ----------------
      #pragma unroll
      for (int i = 0; i < 8; ++i) {
        int row = srow + i * 16;
        int gm  = blockM + row;
        int gk  = kk + skq * 4;
        float4 v;
        v.x = (gm < M && gk + 0 < K) ? A[(size_t)gm * lda + gk + 0] : 0.f;
        v.y = (gm < M && gk + 1 < K) ? A[(size_t)gm * lda + gk + 1] : 0.f;
        v.z = (gm < M && gk + 2 < K) ? A[(size_t)gm * lda + gk + 2] : 0.f;
        v.w = (gm < M && gk + 3 < K) ? A[(size_t)gm * lda + gk + 3] : 0.f;
        store_pk4(&As[row][skq * 4], v);
      }
      #pragma unroll
      for (int i = 0; i < 4; ++i) {
        int n  = srow + i * 16;
        int gn = blockN + n;
        int gk = kk + skq * 4;
        float4 v;
        v.x = (gn < N && gk + 0 < K) ? W[(size_t)gn * K + gk + 0] : 0.f;
        v.y = (gn < N && gk + 1 < K) ? W[(size_t)gn * K + gk + 1] : 0.f;
        v.z = (gn < N && gk + 2 < K) ? W[(size_t)gn * K + gk + 2] : 0.f;
        v.w = (gn < N && gk + 3 < K) ? W[(size_t)gn * K + gk + 3] : 0.f;
        store_pk4(&Bs[n][skq * 4], v);
      }
    }
    __syncthreads();

    // ---- compute: 2 K-substeps x 4 M-frags, B frag reused 4x ----
    const int rb = mt * 64 + (lane & 15);
    const int h  = lane >> 4;
    const int nn = nt * 16 + (lane & 15);
    #pragma unroll
    for (int ks = 0; ks < BK; ks += 32) {
      union { uint4 u[2]; v16bf v; } fb;
      // B: column nn, elements e -> K = ks + 16h + e (16 contiguous bf16)
      fb.u[0] = *(const uint4*)(&Bs[nn][0] + ks + 16 * h);
      fb.u[1] = *(const uint4*)(&Bs[nn][0] + ks + 16 * h + 8);
      #pragma unroll
      for (int f = 0; f < 4; ++f) {
        union { uint4 u[2]; v16bf v; } fa;
        // A: row r, elements e -> K = ks + 16h + e (16 contiguous bf16)
        fa.u[0] = *(const uint4*)(&As[rb + f * 16][0] + ks + 16 * h);
        fa.u[1] = *(const uint4*)(&As[rb + f * 16][0] + ks + 16 * h + 8);
        acc[f] = __builtin_amdgcn_wmma_f32_16x16x32_bf16(
            false, fa.v, false, fb.v, (short)0, acc[f], false, false);
      }
    }
    __syncthreads();
  }

  // ---- epilogue: C/D layout -> vgpr i: lanes0-15 M=i, lanes16-31 M=8+i ----
  const int col = blockN + nt * 16 + (lane & 15);
  const int r00 = blockM + mt * 64 + (lane >> 4) * 8;
  if (col < N) {
    float bv = bias ? bias[col] : 0.f;
    #pragma unroll
    for (int f = 0; f < 4; ++f) {
      #pragma unroll
      for (int i = 0; i < 8; ++i) {
        int row = r00 + f * 16 + i;
        if (row < M) {
          float v = acc[f][i] + bv;
          v = apply_act(v, act);
          if (res) v += res[(size_t)row * ldc + col];
          C[(size_t)row * ldc + col] = v;
        }
      }
    }
  }
}

// ---------------------------------------------------------------------------
// Direct conv1d (NCH, OIH) + optional activation
// ---------------------------------------------------------------------------
__global__ void conv1d_act_kernel(const float* __restrict__ x,
                                  const float* __restrict__ w,
                                  const float* __restrict__ bias,
                                  float* __restrict__ y,
                                  int B, int Cin, int L, int Cout, int Ks,
                                  int stride, int pad, int Lout, int act)
{
  size_t idx = (size_t)blockIdx.x * blockDim.x + threadIdx.x;
  size_t total = (size_t)B * Cout * Lout;
  if (idx >= total) return;
  int lo = (int)(idx % Lout);
  int co = (int)((idx / Lout) % Cout);
  int b  = (int)(idx / ((size_t)Lout * Cout));
  float acc = bias[co];
  int in0 = lo * stride - pad;
  for (int ci = 0; ci < Cin; ++ci) {
    const float* xp = x + ((size_t)b * Cin + ci) * L;
    const float* wp = w + ((size_t)co * Cin + ci) * Ks;
    for (int k = 0; k < Ks; ++k) {
      int p = in0 + k;
      if ((unsigned)p < (unsigned)L) acc += xp[p] * wp[k];
    }
  }
  y[idx] = apply_act(acc, act);
}

// ---------------------------------------------------------------------------
// SAFiLM pieces
// ---------------------------------------------------------------------------
__global__ void downmax_pe_kernel(const float* __restrict__ xc,
                                  float* __restrict__ x0,
                                  int B, int C, int L, int blk)
{
  size_t idx = (size_t)blockIdx.x * blockDim.x + threadIdx.x;
  size_t total = (size_t)B * 64 * C;
  if (idx >= total) return;
  int c = (int)(idx % C);
  int s = (int)((idx / C) % 64);
  int b = (int)(idx / ((size_t)C * 64));
  const float* src = xc + ((size_t)b * C + c) * L + (size_t)s * blk;
  float mx = -3.4e38f;
  for (int j = 0; j < blk; ++j) mx = fmaxf(mx, src[j]);
  int p = c >> 1;
  float ang = (float)s * __expf(-(float)(2 * p) * logf(10000.f) / (float)C);
  float pe = (c & 1) ? cosf(ang) : sinf(ang);
  x0[idx] = mx * sqrtf((float)C) + pe;
}

__global__ void film_kernel(float* __restrict__ xc,
                            const float* __restrict__ x0,
                            int B, int C, int L, int blk)
{
  size_t idx = (size_t)blockIdx.x * blockDim.x + threadIdx.x;
  size_t total = (size_t)B * C * L;
  if (idx >= total) return;
  int l = (int)(idx % L);
  int c = (int)((idx / L) % C);
  int b = (int)(idx / ((size_t)L * C));
  xc[idx] *= x0[((size_t)(b * 64 + l / blk)) * C + c];
}

template <int HD>
__global__ void attention_kernel(const float* __restrict__ qkv,
                                 float* __restrict__ attn, int D)
{
  const int b = blockIdx.x >> 1;
  const int h = blockIdx.x & 1;
  const int s = threadIdx.x;               // 0..63
  __shared__ float sc[64 * 64];
  const float* base = qkv + (size_t)(b * 64) * 3 * D;

  float q[HD];
  #pragma unroll
  for (int d = 0; d < HD; ++d) q[d] = base[(size_t)s * 3 * D + h * HD + d];

  const float scale = rsqrtf((float)HD);
  float mx = -3.4e38f;
  for (int t = 0; t < 64; ++t) {
    const float* kr = base + (size_t)t * 3 * D + D + h * HD;
    float dot = 0.f;
    #pragma unroll
    for (int d = 0; d < HD; ++d) dot += q[d] * kr[d];
    dot *= scale;
    sc[s * 64 + t] = dot;
    mx = fmaxf(mx, dot);
  }
  float sum = 0.f;
  for (int t = 0; t < 64; ++t) {
    float e = __expf(sc[s * 64 + t] - mx);
    sc[s * 64 + t] = e;
    sum += e;
  }
  float inv = 1.f / sum;
  for (int d = 0; d < HD; ++d) {
    float o = 0.f;
    for (int t = 0; t < 64; ++t)
      o += sc[s * 64 + t] * base[(size_t)t * 3 * D + 2 * D + h * HD + d];
    attn[(size_t)(b * 64 + s) * D + h * HD + d] = o * inv;
  }
}

__global__ void layernorm_kernel(const float* __restrict__ x,
                                 const float* __restrict__ g,
                                 const float* __restrict__ bb,
                                 float* __restrict__ out, int M, int D)
{
  int m = blockIdx.x * blockDim.x + threadIdx.x;
  if (m >= M) return;
  const float* r = x + (size_t)m * D;
  float mean = 0.f;
  for (int d = 0; d < D; ++d) mean += r[d];
  mean /= (float)D;
  float var = 0.f;
  for (int d = 0; d < D; ++d) { float t = r[d] - mean; var += t * t; }
  var /= (float)D;
  float inv = rsqrtf(var + 1e-5f);
  float* o = out + (size_t)m * D;
  for (int d = 0; d < D; ++d) o[d] = (r[d] - mean) * inv * g[d] + bb[d];
}

// ---------------------------------------------------------------------------
// Mamba pieces
// ---------------------------------------------------------------------------
__global__ void transpose_bcl_to_blc(const float* __restrict__ x,
                                     float* __restrict__ y, int B, int C, int L)
{
  size_t idx = (size_t)blockIdx.x * blockDim.x + threadIdx.x;
  size_t total = (size_t)B * L * C;
  if (idx >= total) return;
  int c = (int)(idx % C);
  int l = (int)((idx / C) % L);
  int b = (int)(idx / ((size_t)C * L));
  y[idx] = x[((size_t)b * C + c) * L + l];
}

__global__ void transpose_add_blc_to_bcl(const float* __restrict__ m,
                                         const float* __restrict__ x3,
                                         float* __restrict__ y, int B, int C, int L)
{
  size_t idx = (size_t)blockIdx.x * blockDim.x + threadIdx.x;
  size_t total = (size_t)B * C * L;
  if (idx >= total) return;
  int l = (int)(idx % L);
  int c = (int)((idx / L) % C);
  int b = (int)(idx / ((size_t)L * C));
  y[idx] = m[((size_t)b * L + l) * C + c] + x3[idx];
}

__global__ void mamba_dwconv_silu(const float* __restrict__ xz,
                                  const float* __restrict__ cw,
                                  const float* __restrict__ cb,
                                  float* __restrict__ xms, int B, int L)
{
  size_t idx = (size_t)blockIdx.x * blockDim.x + threadIdx.x;
  size_t total = (size_t)B * L * 512;
  if (idx >= total) return;
  int d = (int)(idx % 512);
  int l = (int)((idx / 512) % L);
  int b = (int)(idx / ((size_t)512 * L));
  float acc = cb[d];
  #pragma unroll
  for (int k = 0; k < 4; ++k) {
    int lp = l - 3 + k;
    if (lp >= 0) acc += xz[((size_t)(b * L + lp)) * 1024 + d] * cw[d * 4 + k];
  }
  xms[idx] = acc / (1.f + __expf(-acc));
}

__global__ void mamba_scan_kernel(const float* __restrict__ delta,
                                  const float* __restrict__ xms,
                                  const float* __restrict__ dbl,
                                  const float* __restrict__ A_log,
                                  const float* __restrict__ Dp,
                                  const float* __restrict__ xz,
                                  float* __restrict__ ybuf, int B, int L)
{
  int idx = blockIdx.x * blockDim.x + threadIdx.x;
  if (idx >= B * 512) return;
  int b = idx / 512, d = idx % 512;
  float Aexp[16], h[16];
  #pragma unroll
  for (int n = 0; n < 16; ++n) { Aexp[n] = -__expf(A_log[d * 16 + n]); h[n] = 0.f; }
  float Dv = Dp[d];
  for (int l = 0; l < L; ++l) {
    size_t t = (size_t)b * L + l;
    float dl = delta[t * 512 + d];
    float xv = xms[t * 512 + d];
    const float* Bm = dbl + t * 48 + 16;
    const float* Cm = dbl + t * 48 + 32;
    float y = 0.f;
    #pragma unroll
    for (int n = 0; n < 16; ++n) {
      float dA = __expf(dl * Aexp[n]);
      h[n] = dA * h[n] + dl * Bm[n] * xv;
      y += h[n] * Cm[n];
    }
    y += xv * Dv;
    float zv = xz[t * 1024 + 512 + d];
    ybuf[t * 512 + d] = y * (zv / (1.f + __expf(-zv)));
  }
}

// ---------------------------------------------------------------------------
// misc elementwise
// ---------------------------------------------------------------------------
__global__ void add_inplace_kernel(float* __restrict__ y,
                                   const float* __restrict__ x, size_t n)
{
  size_t i = (size_t)blockIdx.x * blockDim.x + threadIdx.x;
  if (i < n) y[i] += x[i];
}

__global__ void pixel_shuffle_kernel(const float* __restrict__ x,
                                     float* __restrict__ y,
                                     int B, int C, int W, int r)
{
  int C2 = C / r, Lout = W * r;
  size_t idx = (size_t)blockIdx.x * blockDim.x + threadIdx.x;
  size_t total = (size_t)B * C2 * Lout;
  if (idx >= total) return;
  int l  = (int)(idx % Lout);
  int c2 = (int)((idx / Lout) % C2);
  int b  = (int)(idx / ((size_t)Lout * C2));
  int i2 = l % r, wj = l / r;
  y[idx] = x[((size_t)b * C + i2 * C2 + c2) * W + wj];
}

__global__ void tanh_residual_kernel(const float* __restrict__ u,
                                     const float* __restrict__ imu,
                                     float* __restrict__ out, size_t n)
{
  size_t i = (size_t)blockIdx.x * blockDim.x + threadIdx.x;
  if (i < n) out[i] = tanhf(u[i]) + imu[i];
}

// ---------------------------------------------------------------------------
// Host-side orchestration
// ---------------------------------------------------------------------------
static inline int nblk(size_t n) { return (int)((n + 255) / 256); }

static void gemm(hipStream_t st, const float* A, int lda, const float* W,
                 const float* bias, float* C, int ldc, const float* res,
                 int M, int N, int K, int act)
{
  dim3 g((N + BN - 1) / BN, (M + BM - 1) / BM);
  gemm_bf16_wmma<<<g, 256, 0, st>>>(A, lda, W, bias, C, ldc, res, M, N, K, act);
}

struct TLayer { const float* p[12]; };
// p: 0 in_w, 1 in_b, 2 out_w, 3 out_b, 4 ln1_g, 5 ln1_b,
//    6 ff1_w, 7 ff1_b, 8 ff2_w, 9 ff2_b, 10 ln2_g, 11 ln2_b

static void run_safilm(hipStream_t st, float* xc, int B, int C, int L, int blk,
                       const TLayer* Ls, float* arena)
{
  const int D = C, M = B * 64;                 // S = L/blk = 64 for all uses
  float* x0   = arena;
  float* qkv  = x0   + (size_t)M * D;
  float* attn = qkv  + (size_t)M * 3 * D;
  float* tmp  = attn + (size_t)M * D;
  float* ff1  = tmp  + (size_t)M * D;

  downmax_pe_kernel<<<nblk((size_t)M * C), 256, 0, st>>>(xc, x0, B, C, L, blk);
  for (int l = 0; l < 3; ++l) {
    const TLayer& P = Ls[l];
    gemm(st, x0, D, P.p[0], P.p[1], qkv, 3 * D, nullptr, M, 3 * D, D, ACT_NONE);
    if (D == 64) attention_kernel<32><<<B * 2, 64, 0, st>>>(qkv, attn, D);
    else         attention_kernel<64><<<B * 2, 64, 0, st>>>(qkv, attn, D);
    gemm(st, attn, D, P.p[2], P.p[3], tmp, D, /*res=*/x0, M, D, D, ACT_NONE);
    layernorm_kernel<<<nblk(M), 256, 0, st>>>(tmp, P.p[4], P.p[5], x0, M, D);
    gemm(st, x0, D, P.p[6], P.p[7], ff1, 1024, nullptr, M, 1024, D, ACT_RELU);
    gemm(st, ff1, 1024, P.p[8], P.p[9], tmp, D, /*res=*/x0, M, D, 1024, ACT_NONE);
    layernorm_kernel<<<nblk(M), 256, 0, st>>>(tmp, P.p[10], P.p[11], x0, M, D);
  }
  film_kernel<<<nblk((size_t)B * C * L), 256, 0, st>>>(xc, x0, B, C, L, blk);
}

static void load_layers(void* const* d_in, int base, TLayer* L) {
  for (int j = 0; j < 3; ++j)
    for (int q = 0; q < 12; ++q)
      L[j].p[q] = (const float*)d_in[base + j * 12 + q];
}

extern "C" void kernel_launch(void* const* d_in, const int* in_sizes, int n_in,
                              void* d_out, int out_size, void* d_ws, size_t ws_size,
                              hipStream_t stream)
{
  (void)in_sizes; (void)n_in; (void)out_size; (void)ws_size;
  const float* imu   = (const float*)d_in[0];
  const float* dc1_w = (const float*)d_in[1];  const float* dc1_b = (const float*)d_in[2];
  const float* dc2_w = (const float*)d_in[3];  const float* dc2_b = (const float*)d_in[4];
  const float* dc3_w = (const float*)d_in[5];  const float* dc3_b = (const float*)d_in[6];
  const float* uc1_w = (const float*)d_in[7];  const float* uc1_b = (const float*)d_in[8];
  const float* uc2_w = (const float*)d_in[9];  const float* uc2_b = (const float*)d_in[10];
  const float* oc_w  = (const float*)d_in[11]; const float* oc_b  = (const float*)d_in[12];

  TLayer sf1[3], sf2[3], sf1u[3], sf2u[3];
  load_layers(d_in, 13,  sf1);
  load_layers(d_in, 49,  sf2);
  load_layers(d_in, 85,  sf1u);
  load_layers(d_in, 121, sf2u);

  const float* m_in_w   = (const float*)d_in[157];
  const float* m_conv_w = (const float*)d_in[158];
  const float* m_conv_b = (const float*)d_in[159];
  const float* m_xproj  = (const float*)d_in[160];
  const float* m_dt_w   = (const float*)d_in[161];
  const float* m_dt_b   = (const float*)d_in[162];
  const float* m_A_log  = (const float*)d_in[163];
  const float* m_D      = (const float*)d_in[164];
  const float* m_out_w  = (const float*)d_in[165];

  float* ws = (float*)d_ws;
  float* x1    = ws;                    // (64,64,2048)  8388608
  float* x2    = x1 + 8388608;          // (64,128,512)  4194304
  float* x3    = x2 + 4194304;          // (64,256,128)  2097152
  float* bbuf  = x3 + 2097152;          // (64,256,128)  2097152
  float* c1    = bbuf + 2097152;        // (64,128,512)  4194304
  float* c2    = c1 + 4194304;          // (64,64,2048)  8388608
  float* arena = c2 + 8388608;          // transient scratch

  // ---- down path ----
  conv1d_act_kernel<<<nblk((size_t)64 * 64 * 2048), 256, 0, stream>>>(
      imu, dc1_w, dc1_b, x1, 64, 1, 8192, 64, 65, 4, 32, 2048, ACT_LRELU);
  run_safilm(stream, x1, 64, 64, 2048, 32, sf1, arena);

  conv1d_act_kernel<<<nblk((size_t)64 * 128 * 512), 256, 0, stream>>>(
      x1, dc2_w, dc2_b, x2, 64, 64, 2048, 128, 17, 4, 8, 512, ACT_LRELU);
  run_safilm(stream, x2, 64, 128, 512, 8, sf2, arena);

  conv1d_act_kernel<<<nblk((size_t)64 * 256 * 128), 256, 0, stream>>>(
      x2, dc3_w, dc3_b, x3, 64, 128, 512, 256, 7, 4, 3, 128, ACT_LRELU);

  // ---- mamba bottleneck (B=64, L=128, D=256, d_in=512, n=16, dt_rank=16) ----
  {
    const int B = 64, L = 128, M = B * L;        // M = 8192 tokens
    float* xt    = arena;                        // (M,256)  2097152
    float* xz    = xt    + 2097152;              // (M,1024) 8388608
    float* xms   = xz    + 8388608;              // (M,512)  4194304
    float* dbl   = xms   + 4194304;              // (M,48)    393216
    float* delta = dbl   + 393216;               // (M,512)  4194304
    float* ybuf  = delta + 4194304;              // (M,512)  4194304
    float* mout  = ybuf  + 4194304;              // (M,256)  2097152

    transpose_bcl_to_blc<<<nblk((size_t)M * 256), 256, 0, stream>>>(x3, xt, B, 256, L);
    gemm(stream, xt, 256, m_in_w, nullptr, xz, 1024, nullptr, M, 1024, 256, ACT_NONE);
    mamba_dwconv_silu<<<nblk((size_t)M * 512), 256, 0, stream>>>(xz, m_conv_w, m_conv_b, xms, B, L);
    gemm(stream, xms, 512, m_xproj, nullptr, dbl, 48, nullptr, M, 48, 512, ACT_NONE);
    gemm(stream, dbl /* dt = dbl[:, :16] */, 48, m_dt_w, m_dt_b, delta, 512, nullptr,
         M, 512, 16, ACT_SOFTPLUS);
    mamba_scan_kernel<<<nblk((size_t)B * 512), 256, 0, stream>>>(
        delta, xms, dbl, m_A_log, m_D, xz, ybuf, B, L);
    gemm(stream, ybuf, 512, m_out_w, nullptr, mout, 256, nullptr, M, 256, 512, ACT_NONE);
    transpose_add_blc_to_bcl<<<nblk((size_t)64 * 256 * 128), 256, 0, stream>>>(
        mout, x3, bbuf, 64, 256, L);
  }

  // ---- up path ----
  {
    float* u1pre = arena;                        // (64,512,128) 4194304
    conv1d_act_kernel<<<nblk((size_t)64 * 512 * 128), 256, 0, stream>>>(
        bbuf, uc1_w, uc1_b, u1pre, 64, 256, 128, 512, 7, 1, 3, 128, ACT_LRELU);
    pixel_shuffle_kernel<<<nblk((size_t)64 * 128 * 512), 256, 0, stream>>>(
        u1pre, c1, 64, 512, 128, 4);
  }
  run_safilm(stream, c1, 64, 128, 512, 8, sf1u, arena);
  add_inplace_kernel<<<nblk((size_t)4194304), 256, 0, stream>>>(c1, x2, 4194304);

  {
    float* u2pre = arena;                        // (64,256,512) 8388608
    conv1d_act_kernel<<<nblk((size_t)64 * 256 * 512), 256, 0, stream>>>(
        c1, uc2_w, uc2_b, u2pre, 64, 128, 512, 256, 17, 1, 8, 512, ACT_LRELU);
    pixel_shuffle_kernel<<<nblk((size_t)64 * 64 * 2048), 256, 0, stream>>>(
        u2pre, c2, 64, 256, 512, 4);
  }
  run_safilm(stream, c2, 64, 64, 2048, 32, sf2u, arena);
  add_inplace_kernel<<<nblk((size_t)8388608), 256, 0, stream>>>(c2, x1, 8388608);

  {
    float* u3pre  = arena;                       // (64,4,2048) 524288
    float* u3post = u3pre + 524288;              // (64,1,8192) 524288
    conv1d_act_kernel<<<nblk((size_t)64 * 4 * 2048), 256, 0, stream>>>(
        c2, oc_w, oc_b, u3pre, 64, 64, 2048, 4, 65, 1, 32, 2048, ACT_NONE);
    pixel_shuffle_kernel<<<nblk((size_t)64 * 8192), 256, 0, stream>>>(
        u3pre, u3post, 64, 4, 2048, 4);
    tanh_residual_kernel<<<nblk((size_t)524288), 256, 0, stream>>>(
        u3post, imu, (float*)d_out, 524288);
  }
}
